// ExpanderLinearLayer_43920335569022
// MI455X (gfx1250) — compile-verified, matrix operand order checked by reading, not run on hardware
//
#include <hip/hip_runtime.h>

// C = X @ (W * mask)^T
// X: [16384, 2048] f32, W/mask: [2048, 2048] f32/i32, C: [16384, 2048] f32
// Path: fp32 -> bf16 (round-to-nearest via +0x8000 + v_perm_b32 pack) at LDS
// staging, mask folded bitwise into W, v_wmma_f32_16x16x32_bf16, f32 accum.
// Double-buffered LDS, one barrier per K-step, launch_bounds(256,1) so the
// 48-VGPR global staging set stays in registers (no scratch spills).

#define GEMM_M   16384
#define GEMM_K   2048
#define GEMM_O   2048

#define TILE_M   128
#define TILE_N   128
#define TILE_K   32
#define LDS_STRIDE 80                 // 32 bf16 (64B) + 16B pad; 16B-aligned rows
#define LDS_TILE   (TILE_M * LDS_STRIDE)

typedef __attribute__((ext_vector_type(16))) __bf16 v16bf;
typedef __attribute__((ext_vector_type(8)))  float  v8f;

union FragBF16 {
    uint4 q[2];   // 32 bytes
    v16bf v;
};

// Pack bf16(a),bf16(b) -> dword. Round-to-nearest via +0x8000, then one
// v_perm_b32 selects the two high halves: D = {b[31:16], a[31:16]}.
__device__ __forceinline__ unsigned int pack2_bf16(unsigned int ua, unsigned int ub) {
    return __builtin_amdgcn_perm(ub + 0x8000u, ua + 0x8000u, 0x07060302u);
}

__global__ __launch_bounds__(256, 1)
void expander_linear_wmma_bf16(const float* __restrict__ X,
                               const float* __restrict__ W,
                               const int*   __restrict__ Mask,
                               float*       __restrict__ Out) {
    __shared__ __align__(16) unsigned char ldsA[2][LDS_TILE]; // X tile, bf16
    __shared__ __align__(16) unsigned char ldsB[2][LDS_TILE]; // masked W tile, bf16

    const int tid   = threadIdx.x;
    const int lane  = tid & 31;
    const int wave  = tid >> 5;
    const int waveM = wave & 3;   // 4 waves along M: 32-row bands
    const int waveN = wave >> 2;  // 2 waves along N: 64-col bands

    const int blockN = blockIdx.x;  // OUT tiles (16)
    const int blockM = blockIdx.y;  // M tiles  (128)

    // -------- global->LDS staging: each thread moves 16 elements of X and W
    const int ldRow = tid >> 1;          // 0..127
    const int ldCol = (tid & 1) * 16;    // 0 or 16 (within 32-wide K chunk)

    const size_t xBase = (size_t)(blockM * TILE_M + ldRow) * GEMM_K + ldCol;
    const size_t wBase = (size_t)(blockN * TILE_N + ldRow) * GEMM_K + ldCol;

    float4 ax[4];   // 16 floats of X
    float4 wx[4];   // 16 floats of W
    int4   mx[4];   // 16 ints of mask

    auto load_stage = [&](int kk) {
        const float4* xp = (const float4*)(X    + xBase + kk);
        const float4* wp = (const float4*)(W    + wBase + kk);
        const int4*   mp = (const int4*)  (Mask + wBase + kk);
#pragma unroll
        for (int i = 0; i < 4; ++i) { ax[i] = xp[i]; wx[i] = wp[i]; mx[i] = mp[i]; }
        // speculative prefetch of the chunk after next (warms L2/L0)
        __builtin_prefetch(X    + xBase + kk + 2 * TILE_K, 0, 1);
        __builtin_prefetch(W    + wBase + kk + 2 * TILE_K, 0, 1);
        __builtin_prefetch(Mask + wBase + kk + 2 * TILE_K, 0, 1);
    };

    auto store_stage = [&](int p) {
        const float* af = (const float*)ax;
        const float* wf = (const float*)wx;
        const int*   mf = (const int*)  mx;
        unsigned int pa[8], pb[8];
#pragma unroll
        for (int i = 0; i < 8; ++i) {
            pa[i] = pack2_bf16(__float_as_uint(af[2 * i]),
                               __float_as_uint(af[2 * i + 1]));
            // mask in {0,1}: w & (0-m) zeroes masked-out weights, no branches
            unsigned int w0 = __float_as_uint(wf[2 * i])     & (unsigned int)(0 - mf[2 * i]);
            unsigned int w1 = __float_as_uint(wf[2 * i + 1]) & (unsigned int)(0 - mf[2 * i + 1]);
            pb[i] = pack2_bf16(w0, w1);
        }
        uint4* da = (uint4*)(ldsA[p] + ldRow * LDS_STRIDE + ldCol * 2);
        uint4* db = (uint4*)(ldsB[p] + ldRow * LDS_STRIDE + ldCol * 2);
        da[0] = uint4{pa[0], pa[1], pa[2], pa[3]};
        da[1] = uint4{pa[4], pa[5], pa[6], pa[7]};
        db[0] = uint4{pb[0], pb[1], pb[2], pb[3]};
        db[1] = uint4{pb[4], pb[5], pb[6], pb[7]};
    };

    // -------- WMMA fragment addressing (per ISA 16-bit A/B layout):
    // lane 0-15 : row = lane,    K = [0..7] then [16..23]
    // lane 16-31: row = lane-16, K = [8..15] then [24..31]
    const int rowSel = lane & 15;
    const int kOff   = (lane >> 4) * 8;            // in bf16 elements
    const int aRow0  = waveM * 32 + rowSel;
    const int bRow0  = waveN * 64 + rowSel;

    v8f acc[2][4] = {};   // zero-initialized accumulators

    // -------- software pipeline: regs hold chunk kk+32, LDS buf p holds chunk kk
    load_stage(0);
    store_stage(0);
    __syncthreads();

    int p = 0;
    for (int kk = 0; kk < GEMM_K; kk += TILE_K) {
        const bool hasNext = (kk + TILE_K) < GEMM_K;
        if (hasNext)
            load_stage(kk + TILE_K);   // global loads overlap WMMA below

        FragBF16 afrag[2], bfrag[4];
#pragma unroll
        for (int mi = 0; mi < 2; ++mi) {
            const unsigned char* pp = ldsA[p] + (aRow0 + mi * 16) * LDS_STRIDE + kOff * 2;
            afrag[mi].q[0] = *(const uint4*)(pp);
            afrag[mi].q[1] = *(const uint4*)(pp + 32);
        }
#pragma unroll
        for (int ni = 0; ni < 4; ++ni) {
            const unsigned char* pp = ldsB[p] + (bRow0 + ni * 16) * LDS_STRIDE + kOff * 2;
            bfrag[ni].q[0] = *(const uint4*)(pp);
            bfrag[ni].q[1] = *(const uint4*)(pp + 32);
        }
#pragma unroll
        for (int mi = 0; mi < 2; ++mi)
#pragma unroll
            for (int ni = 0; ni < 4; ++ni)
                acc[mi][ni] = __builtin_amdgcn_wmma_f32_16x16x32_bf16(
                    false, afrag[mi].v, false, bfrag[ni].v,
                    (short)0, acc[mi][ni], false, false);

        if (hasNext)
            store_stage(p ^ 1);        // fill the other buffer (no reader conflict)

        __syncthreads();               // single barrier per K-step
        p ^= 1;
    }

    // -------- epilogue: C/D layout -> VGPR r holds (M = (lane>>4)*8 + r, N = lane&15)
    const int outRowBase = blockM * TILE_M + waveM * 32 + (lane >> 4) * 8;
    const int outColBase = blockN * TILE_N + waveN * 64 + (lane & 15);
#pragma unroll
    for (int mi = 0; mi < 2; ++mi) {
#pragma unroll
        for (int ni = 0; ni < 4; ++ni) {
            float* op = Out + (size_t)(outRowBase + mi * 16) * GEMM_O + outColBase + ni * 16;
#pragma unroll
            for (int r = 0; r < 8; ++r)
                op[(size_t)r * GEMM_O] = acc[mi][ni][r];
        }
    }
}

extern "C" void kernel_launch(void* const* d_in, const int* in_sizes, int n_in,
                              void* d_out, int out_size, void* d_ws, size_t ws_size,
                              hipStream_t stream) {
    const float* X    = (const float*)d_in[0];
    const float* W    = (const float*)d_in[1];
    const int*   Mask = (const int*)d_in[2];
    float*       Out  = (float*)d_out;

    dim3 grid(GEMM_O / TILE_N, GEMM_M / TILE_M);  // (16, 128)
    expander_linear_wmma_bf16<<<grid, 256, 0, stream>>>(X, W, Mask, Out);
}
